// SparseAttention_13280038879366
// MI455X (gfx1250) — matrix operand order repaired and checked by reading, not compiled, not run on hardware
//
#include <hip/hip_runtime.h>
#include <hip/hip_bf16.h>

// ---------------------------------------------------------------------------
// Sparse attention for MI455X (gfx1250, wave32, WMMA).
// Pipeline: f32->bf16 casts; bf16 WMMA QKV GEMM with ASYNC global->LDS tile
// staging (GLOBAL_LOAD_ASYNC_TO_LDS_B128 / s_wait_asynccnt); fused
// per-(b,h,16q) attention with 128KB LDS score tile + exact radix-select
// top-k threshold + softmax + WMMA P*V; bf16 WMMA out-projection.
// ---------------------------------------------------------------------------

#define B_SZ   2
#define L_SZ   2048
#define D_SZ   1024
#define H_SZ   16
#define DH     64
#define KSEL   204          // int(L * (1 - 0.9))
#define QKV_LD 3072

typedef __bf16 v16bf __attribute__((ext_vector_type(16)));
typedef float  f32x8 __attribute__((ext_vector_type(8)));
typedef int    v4i   __attribute__((ext_vector_type(4)));

union Frag { v16bf v; unsigned u[8]; };

#if defined(__gfx1250__) && __has_builtin(__builtin_amdgcn_global_load_async_to_lds_b128)
#define HAVE_ASYNC_LDS 1
#else
#define HAVE_ASYNC_LDS 0
#endif

#if HAVE_ASYNC_LDS
typedef __attribute__((address_space(1))) v4i g_v4i;   // global int4
typedef __attribute__((address_space(3))) v4i l_v4i;   // LDS int4
#endif

// 16-byte global -> LDS copy; async (no VGPR round-trip, ASYNCcnt) if available
static __device__ __forceinline__ void copy16_g2l(void* lds_dst, const void* gsrc) {
#if HAVE_ASYNC_LDS
    __builtin_amdgcn_global_load_async_to_lds_b128(
        (g_v4i*)(uintptr_t)gsrc,
        (l_v4i*)(unsigned)(uintptr_t)lds_dst, 0, 0);
#else
    *(uint4*)lds_dst = *(const uint4*)gsrc;
#endif
}

static __device__ __forceinline__ void wait_async_copies() {
#if HAVE_ASYNC_LDS
#if __has_builtin(__builtin_amdgcn_s_wait_asynccnt)
    __builtin_amdgcn_s_wait_asynccnt(0);
#else
    asm volatile("s_wait_asynccnt 0" ::: "memory");
#endif
#endif
}

static __device__ __forceinline__ unsigned short f2bf(float f) {
    unsigned u = __float_as_uint(f);
    unsigned r = u + 0x7FFFu + ((u >> 16) & 1u);   // round-to-nearest-even
    return (unsigned short)(r >> 16);
}

static __device__ __forceinline__ unsigned fkey(float f) {
    // order-preserving mapping: larger float -> larger unsigned key
    unsigned u = __float_as_uint(f);
    return u ^ ((u >> 31) ? 0xFFFFFFFFu : 0x80000000u);
}
static __device__ __forceinline__ float fkey_inv(unsigned k) {
    unsigned u = (k >> 31) ? (k ^ 0x80000000u) : ~k;
    return __uint_as_float(u);
}

static __device__ __forceinline__ f32x8 wmma_bf16(const Frag& a, const Frag& b, f32x8 c) {
    return __builtin_amdgcn_wmma_f32_16x16x32_bf16(
        false, a.v, false, b.v, (short)0, c, false, false);
}

// ---------------------------------------------------------------------------
// f32 -> bf16, 4 elements / thread
// ---------------------------------------------------------------------------
__global__ __launch_bounds__(256) void f32_to_bf16_x4(
    const float* __restrict__ s, unsigned short* __restrict__ d, int n4)
{
    int i = blockIdx.x * 256 + threadIdx.x;
    if (i >= n4) return;
    float4 f = ((const float4*)s)[i];
    uint2 o;
    o.x = (unsigned)f2bf(f.x) | ((unsigned)f2bf(f.y) << 16);
    o.y = (unsigned)f2bf(f.z) | ((unsigned)f2bf(f.w) << 16);
    ((uint2*)d)[i] = o;
}

// ---------------------------------------------------------------------------
// C[m,n] = sum_k A[m,k]*B[n,k] + bias[n]   (A: MxK bf16 rm, B: NxK bf16 rm)
// Tile: 64(M) x 128(N), 256 thr = 8 waves in 4x2; K staged in LDS chunks of 32
// via async global->LDS copies.  Output bf16 (Cbf) or f32 (Cf), ld = ldc.
// ---------------------------------------------------------------------------
__global__ __launch_bounds__(256) void gemm_bf16_nt(
    const unsigned short* __restrict__ A, const unsigned short* __restrict__ Bm,
    const float* __restrict__ bias, unsigned short* __restrict__ Cbf,
    float* __restrict__ Cf, int M, int N, int K, int ldc)
{
    __shared__ __align__(16) unsigned short As[64 * 32];
    __shared__ __align__(16) unsigned short Bs[128 * 32];
    const int t = threadIdx.x, lane = t & 31, wid = t >> 5;
    const int m0 = blockIdx.y * 64, n0 = blockIdx.x * 128;
    const int wm = (wid & 3) * 16, wn = (wid >> 2) * 64;

    f32x8 acc[4];
#pragma unroll
    for (int nt = 0; nt < 4; nt++)
#pragma unroll
        for (int r = 0; r < 8; r++) acc[nt][r] = 0.f;

    const int kb  = (lane < 16) ? 0 : 8;
    const int kb2 = (lane < 16) ? 0 : 16;

    // per-thread staging coordinates (A: 8 bf16, B: 16 bf16 per thread)
    const int ar = t >> 2,  ac = (t & 3) * 8;
    const int br = t >> 1,  bc = (t & 1) * 16;

    for (int k0 = 0; k0 < K; k0 += 32) {
        __syncthreads();     // previous chunk fully consumed
        copy16_g2l(&As[ar * 32 + ac],     &A[(size_t)(m0 + ar) * K + k0 + ac]);
        copy16_g2l(&Bs[br * 32 + bc],     &Bm[(size_t)(n0 + br) * K + k0 + bc]);
        copy16_g2l(&Bs[br * 32 + bc + 8], &Bm[(size_t)(n0 + br) * K + k0 + bc + 8]);
        wait_async_copies();
        __syncthreads();

        Frag fa;
        const int am = wm + (lane & 15);
#pragma unroll
        for (int j = 0; j < 8; j++) {
            int kk = kb2 + 2 * j;
            fa.u[j] = *(const unsigned*)&As[am * 32 + kk];
        }
#pragma unroll
        for (int nt = 0; nt < 4; nt++) {
            Frag fb;
            const int bn = wn + nt * 16 + (lane & 15);
#pragma unroll
            for (int j = 0; j < 8; j++)
                fb.u[j] = *(const unsigned*)&Bs[bn * 32 + kb2 + 2 * j];
            acc[nt] = wmma_bf16(fa, fb, acc[nt]);
        }
    }

#pragma unroll
    for (int nt = 0; nt < 4; nt++) {
        const int col = n0 + wn + nt * 16 + (lane & 15);
        const float bv = bias[col];
#pragma unroll
        for (int r = 0; r < 8; r++) {
            const int row = m0 + wm + r + ((lane >> 4) << 3);
            const float v = acc[nt][r] + bv;
            if (Cf) Cf[(size_t)row * ldc + col] = v;
            else    Cbf[(size_t)row * ldc + col] = f2bf(v);
        }
    }
}

// ---------------------------------------------------------------------------
// One workgroup = (b, h, 16 queries). 256 thr = 8 waves.
// LDS: Ss[16][2048] f32 scores/probs (128KB) + Obuf[16][64] + hist[8][256].
// ---------------------------------------------------------------------------
__global__ __launch_bounds__(256) void sparse_attn_kernel(
    const unsigned short* __restrict__ qkv, float* __restrict__ probs_out,
    unsigned short* __restrict__ attn_out)
{
    extern __shared__ char smem_raw[];
    float*    Ss   = (float*)smem_raw;            // 16*2048
    float*    Obuf = Ss + 16 * L_SZ;              // 16*64
    unsigned* hist = (unsigned*)(Obuf + 16 * DH); // 8*256

    const int qt = blockIdx.x, h = blockIdx.y, b = blockIdx.z;
    const int t = threadIdx.x, lane = t & 31, wid = t >> 5;
    const int qrow0 = qt * 16;
    const int kb  = (lane < 16) ? 0 : 8;
    const int kb2 = (lane < 16) ? 0 : 16;

    // ---- Q fragments (16 x 64, two K=32 chunks) ----
    Frag aQ[2];
    {
        const int m = qrow0 + (lane & 15);
        const size_t grow = (size_t)(b * L_SZ + m) * QKV_LD + h * DH;
#pragma unroll
        for (int c = 0; c < 2; c++)
#pragma unroll
            for (int j = 0; j < 8; j++) {
                int d = c * 32 + kb2 + 2 * j;
                aQ[c].u[j] = *(const unsigned*)&qkv[grow + d];
            }
    }

    // ---- Phase 1: scores = Q K^T * scale into LDS ----
    const float scale = 0.125f;   // 1/sqrt(64)
    for (int kt = wid; kt < L_SZ / 16; kt += 8) {
        f32x8 s;
#pragma unroll
        for (int r = 0; r < 8; r++) s[r] = 0.f;
        const int kk = kt * 16 + (lane & 15);
        const size_t grow = (size_t)(b * L_SZ + kk) * QKV_LD + D_SZ + h * DH;
        Frag fb0, fb1;
#pragma unroll
        for (int j = 0; j < 8; j++) {
            fb0.u[j] = *(const unsigned*)&qkv[grow + kb2 + 2 * j];
            fb1.u[j] = *(const unsigned*)&qkv[grow + 32 + kb2 + 2 * j];
        }
        s = wmma_bf16(aQ[0], fb0, s);
        s = wmma_bf16(aQ[1], fb1, s);
#pragma unroll
        for (int r = 0; r < 8; r++) {
            int row = r + ((lane >> 4) << 3);
            Ss[row * L_SZ + kt * 16 + (lane & 15)] = s[r] * scale;
        }
    }
    __syncthreads();

    // ---- Phase 2: per-row exact kth-largest + masked softmax + write probs ----
    for (int rr = 0; rr < 2; rr++) {
        const int row = wid * 2 + rr;
        float* srow = Ss + row * L_SZ;

        float m = -__builtin_inff();
        for (int i = lane; i < L_SZ; i += 32) m = fmaxf(m, srow[i]);
#pragma unroll
        for (int off = 16; off > 0; off >>= 1) m = fmaxf(m, __shfl_xor(m, off, 32));

        unsigned prefix = 0; int kneed = KSEL;
        unsigned* hw = hist + wid * 256;
        for (int p = 0; p < 4; p++) {
            const int shift = 24 - 8 * p;
            for (int i = lane; i < 256; i += 32) hw[i] = 0;
            __syncthreads();
            for (int i = lane; i < L_SZ; i += 32) {
                unsigned key = fkey(srow[i]);
                unsigned hi = 0;
                if (p) hi = (key ^ prefix) >> (shift + 8);
                if (hi == 0) atomicAdd(&hw[(key >> shift) & 255u], 1u);
            }
            __syncthreads();
            unsigned c = 0; int digit = 0;       // all lanes scan identically
            for (int dg = 255; dg >= 0; dg--) {
                unsigned hh = hw[dg];
                if (c + hh >= (unsigned)kneed) { digit = dg; kneed -= (int)c; break; }
                c += hh;
            }
            prefix |= ((unsigned)digit) << shift;
            __syncthreads();
        }
        const float thr = fkey_inv(prefix);

        float sum = 0.f;
        for (int i = lane; i < L_SZ; i += 32) {
            float v  = srow[i];
            float pv = (v >= thr) ? __expf(v - m) : 0.f;
            srow[i] = pv;
            sum += pv;
        }
#pragma unroll
        for (int off = 16; off > 0; off >>= 1) sum += __shfl_xor(sum, off, 32);
        const float inv = 1.f / sum;
        const size_t pbase = ((size_t)((b * H_SZ + h) * L_SZ + qrow0 + row)) * L_SZ;
        for (int i = lane; i < L_SZ; i += 32) {
            float pv = srow[i] * inv;
            srow[i] = pv;
            probs_out[pbase + i] = pv;
        }
    }
    __syncthreads();

    // ---- Phase 3: out = P (16x2048) * V (2048x64) ----
    for (int i = t; i < 16 * DH; i += 256) Obuf[i] = 0.f;
    __syncthreads();
    {
        f32x8 acc[4];
#pragma unroll
        for (int nt = 0; nt < 4; nt++)
#pragma unroll
            for (int r = 0; r < 8; r++) acc[nt][r] = 0.f;

        const int am = lane & 15;
        for (int c = wid * 8; c < wid * 8 + 8; c++) {   // 32 keys per chunk
            const int kk0 = c * 32;
            Frag fa;
#pragma unroll
            for (int j = 0; j < 8; j++) {
                int ko = kb2 + 2 * j;
                float2 pv = *(const float2*)&Ss[am * L_SZ + kk0 + ko];
                fa.u[j] = (unsigned)f2bf(pv.x) | ((unsigned)f2bf(pv.y) << 16);
            }
#pragma unroll
            for (int nt = 0; nt < 4; nt++) {
                Frag fb;
                const int d = nt * 16 + (lane & 15);
#pragma unroll
                for (int j = 0; j < 8; j++) {
                    size_t r0 = (size_t)(b * L_SZ + kk0 + kb2 + 2 * j) * QKV_LD + 2 * D_SZ + h * DH + d;
                    unsigned lo = qkv[r0];
                    unsigned hi2 = qkv[r0 + QKV_LD];
                    fb.u[j] = lo | (hi2 << 16);
                }
                acc[nt] = wmma_bf16(fa, fb, acc[nt]);
            }
        }
#pragma unroll
        for (int nt = 0; nt < 4; nt++)
#pragma unroll
            for (int r = 0; r < 8; r++) {
                int row = r + ((lane >> 4) << 3);
                int col = nt * 16 + (lane & 15);
                atomicAdd(&Obuf[row * DH + col], acc[nt][r]);
            }
    }
    __syncthreads();
    for (int i = t; i < 16 * DH; i += 256) {
        int row = i >> 6, col = i & 63;
        attn_out[(size_t)(b * L_SZ + qrow0 + row) * D_SZ + h * DH + col] = f2bf(Obuf[i]);
    }
}

// ---------------------------------------------------------------------------
extern "C" void kernel_launch(void* const* d_in, const int* in_sizes, int n_in,
                              void* d_out, int out_size, void* d_ws, size_t ws_size,
                              hipStream_t stream)
{
    (void)in_sizes; (void)n_in; (void)out_size; (void)ws_size;
    const float* x    = (const float*)d_in[0];
    const float* Wqkv = (const float*)d_in[1];
    const float* bqkv = (const float*)d_in[2];
    const float* Wout = (const float*)d_in[3];
    const float* bout = (const float*)d_in[4];

    float* out   = (float*)d_out;
    float* probs = out + (size_t)B_SZ * L_SZ * D_SZ;

    char* ws = (char*)d_ws;   // 48 MB used
    unsigned short* x_bf    = (unsigned short*)(ws);               //  8 MB
    unsigned short* wqkv_bf = (unsigned short*)(ws + 8388608);     //  6 MB
    unsigned short* wout_bf = (unsigned short*)(ws + 14680064);    //  2 MB
    unsigned short* qkv_bf  = (unsigned short*)(ws + 16777216);    // 24 MB
    unsigned short* ao_bf   = (unsigned short*)(ws + 41943040);    //  8 MB

    f32_to_bf16_x4<<<(B_SZ * L_SZ * D_SZ / 4) / 256, 256, 0, stream>>>(x, x_bf, B_SZ * L_SZ * D_SZ / 4);
    f32_to_bf16_x4<<<(3 * D_SZ * D_SZ / 4) / 256, 256, 0, stream>>>(Wqkv, wqkv_bf, 3 * D_SZ * D_SZ / 4);
    f32_to_bf16_x4<<<(D_SZ * D_SZ / 4) / 256, 256, 0, stream>>>(Wout, wout_bf, D_SZ * D_SZ / 4);

    // QKV projection: (4096 x 1024) x (3072 x 1024)^T -> bf16 qkv
    gemm_bf16_nt<<<dim3(QKV_LD / 128, (B_SZ * L_SZ) / 64), 256, 0, stream>>>(
        x_bf, wqkv_bf, bqkv, qkv_bf, nullptr, B_SZ * L_SZ, QKV_LD, D_SZ, QKV_LD);

    // Fused sparse attention
    size_t smem = (size_t)(16 * L_SZ + 16 * DH) * sizeof(float) + 8 * 256 * sizeof(unsigned);
    sparse_attn_kernel<<<dim3(L_SZ / 16, H_SZ, B_SZ), 256, smem, stream>>>(qkv_bf, probs, ao_bf);

    // Output projection: (4096 x 1024) x (1024 x 1024)^T -> f32 d_out
    gemm_bf16_nt<<<dim3(D_SZ / 128, (B_SZ * L_SZ) / 64), 256, 0, stream>>>(
        ao_bf, wout_bf, bout, nullptr, out, B_SZ * L_SZ, D_SZ, D_SZ, D_SZ);
}
